// CostVolume2D_30897994728139
// MI455X (gfx1250) — matrix operand order repaired and checked by reading, not compiled, not run on hardware
//
#include <hip/hip_runtime.h>
#include <stdint.h>

typedef __attribute__((ext_vector_type(16))) __bf16       v16bf;
typedef __attribute__((ext_vector_type(8)))  float        v8f;
typedef __attribute__((ext_vector_type(8)))  unsigned int v8u;

#define NB   4
#define CC   32
#define HH   256
#define WW   512
#define DMAX 48
#define CH   (HH * WW)   // channel stride in elements

// Pack the upper 16 bits (truncate-to-bf16) of two f32 bit patterns into one
// dword: [bf16(x0) low | bf16(x1) high] via one v_perm_b32.
__device__ __forceinline__ unsigned int pack_hi16(unsigned int u0, unsigned int u1) {
  return __builtin_amdgcn_perm(u1, u0, 0x07060302u);
}

// Split two f32 into packed (hi, lo) bf16 dwords.
// hi = trunc_bf16(x); x - hi is EXACT in f32; lo = trunc_bf16(x - hi).
// Total representation error ~2^-17 relative.
__device__ __forceinline__ void split2(float x0, float x1,
                                       unsigned int& hi, unsigned int& lo) {
  const unsigned int u0 = __builtin_bit_cast(unsigned int, x0);
  const unsigned int u1 = __builtin_bit_cast(unsigned int, x1);
  const float h0 = __builtin_bit_cast(float, u0 & 0xFFFF0000u);
  const float h1 = __builtin_bit_cast(float, u1 & 0xFFFF0000u);
  const float l0 = x0 - h0;
  const float l1 = x1 - h1;
  hi = pack_hi16(u0, u1);
  lo = pack_hi16(__builtin_bit_cast(unsigned int, l0),
                 __builtin_bit_cast(unsigned int, l1));
}

// Load one shifted A (R) tile (M x K = 16 x 32), split to bf16 hi/lo, and do
// the 3-WMMA "bf16x2" product against the resident B tile.
// A layout: m = lane%16; lanes 0-15: VGPR0-3 K=0..7, VGPR4-7 K=16..23;
// lanes 16-31: VGPR0-3 K=8..15, VGPR4-7 K=24..31 (2 per VGPR, low first).
template <bool GUARD>
__device__ __forceinline__ v8f mma_pair(const float* __restrict__ Rrow,
                                        int wr, int g,
                                        v16bf b_hi, v16bf b_lo) {
  v8u ahu, alu;
#pragma unroll
  for (int v = 0; v < 8; ++v) {
    const int c0 = (v < 4) ? (8 * g + 2 * v) : (16 + 8 * g + 2 * (v - 4));
    float x0 = 0.0f, x1 = 0.0f;
    if (!GUARD || wr >= 0) {               // left zero-pad only near w=0
      x0 = Rrow[(size_t)c0 * CH + wr];
      x1 = Rrow[(size_t)(c0 + 1) * CH + wr];
    }
    unsigned int hi, lo;
    split2(x0, x1, hi, lo);
    ahu[v] = hi; alu[v] = lo;
  }
  const v16bf a_hi = __builtin_bit_cast(v16bf, ahu);
  const v16bf a_lo = __builtin_bit_cast(v16bf, alu);

  v8f acc = {};
  acc = __builtin_amdgcn_wmma_f32_16x16x32_bf16(false, a_hi, false, b_hi,
                                                (short)0, acc, false, false);
  acc = __builtin_amdgcn_wmma_f32_16x16x32_bf16(false, a_hi, false, b_lo,
                                                (short)0, acc, false, false);
  acc = __builtin_amdgcn_wmma_f32_16x16x32_bf16(false, a_lo, false, b_hi,
                                                (short)0, acc, false, false);
  return acc;
}

template <bool BORDER>
__device__ __forceinline__ void do_tile(const float* __restrict__ Lrow,
                                        const float* __restrict__ Rrow,
                                        float* __restrict__ Orow,
                                        int w0, int lane) {
  const int g   = lane >> 4;   // half-wave group
  const int col = lane & 15;   // N column (w) for B/D; M row (w') for A

  // ---- B matrix (K x N = 32 x 16) from L tile, pre-scaled by 1/CC ----
  // lanes 0-15: VGPR0-3 K=0..7, VGPR4-7 K=16..23; lanes 16-31: VGPR0-3 K=8..15, VGPR4-7 K=24..31.
  v8u bh, bl;
  {
    const int wB = w0 + col;
#pragma unroll
    for (int v = 0; v < 8; ++v) {
      const int c0 = (v < 4) ? (8 * g + 2 * v) : (16 + 8 * g + 2 * (v - 4));
      const float x0 = Lrow[(size_t)c0 * CH + wB] * (1.0f / (float)CC);
      const float x1 = Lrow[(size_t)(c0 + 1) * CH + wB] * (1.0f / (float)CC);
      unsigned int hi, lo;
      split2(x0, x1, hi, lo);
      bh[v] = hi; bl[v] = lo;
    }
  }
  const v16bf b_hi = __builtin_bit_cast(v16bf, bh);
  const v16bf b_lo = __builtin_bit_cast(v16bf, bl);

  float* const Ocol = Orow + (w0 + col);   // per-lane output column base

  // kk=0: wr = w0+col >= 0 always -> never needs the load guard.
  const v8f acc0 = mma_pair<false>(Rrow, w0 + col, g, b_hi, b_lo);

  // kk=1, kk=2: d = 16k + col - m is always in [0,48) -> unconditional stores.
#pragma unroll
  for (int kk = 1; kk <= 2; ++kk) {
    const v8f acc = mma_pair<BORDER>(Rrow, w0 - 16 * kk + col, g, b_hi, b_lo);
#pragma unroll
    for (int v = 0; v < 8; ++v) {
      const int d = 16 * kk + col - (v + 8 * g);   // [1,31] / [17,47]
      Ocol[(size_t)d * CH] = acc[v];
    }
  }

  // kk=3, merged with kk=0: validity masks are exact complements.
  // col >= m -> kk=0 valid (d = col-m in [0,15]); col < m -> kk=3 valid
  // (d = 48+col-m in [33,47]). One unconditional store per v, EXEC untouched.
  const v8f acc3 = mma_pair<BORDER>(Rrow, w0 - 48 + col, g, b_hi, b_lo);
#pragma unroll
  for (int v = 0; v < 8; ++v) {
    const int  m    = v + 8 * g;
    const bool sel0 = (col >= m);
    const int  d    = sel0 ? (col - m) : (48 + col - m);
    const float val = sel0 ? acc0[v] : acc3[v];
    Ocol[(size_t)d * CH] = val;
  }
}

// One wave per (n, h, 16-wide w tile); 8 waves per 256-thread block.
__global__ __launch_bounds__(256)
void cost_volume_wmma_bf16x2(const float* __restrict__ L,
                             const float* __restrict__ R,
                             float* __restrict__ out) {
  const int lane = threadIdx.x & 31;
  const int wave = threadIdx.x >> 5;
  const int t    = blockIdx.x * 8 + wave;   // tile id over N*H*(W/16) = 32768
  const int wt   = t & 31;                  // W/16 = 32 tiles per row
  const int h    = (t >> 5) & 255;          // H = 256
  const int n    = t >> 13;
  const int w0   = wt << 4;

  const float* Lrow = L   + (size_t)n * CC * CH   + (size_t)h * WW;
  const float* Rrow = R   + (size_t)n * CC * CH   + (size_t)h * WW;
  float*       Orow = out + (size_t)n * DMAX * CH + (size_t)h * WW;

  // Wave-uniform branch: interior tiles (w0 >= 48) never hit the zero-pad.
  if (w0 >= DMAX) do_tile<false>(Lrow, Rrow, Orow, w0, lane);
  else            do_tile<true >(Lrow, Rrow, Orow, w0, lane);
}

extern "C" void kernel_launch(void* const* d_in, const int* in_sizes, int n_in,
                              void* d_out, int out_size, void* d_ws, size_t ws_size,
                              hipStream_t stream) {
  (void)in_sizes; (void)n_in; (void)out_size; (void)d_ws; (void)ws_size;
  const float* l_fmap = (const float*)d_in[0];
  const float* r_fmap = (const float*)d_in[1];
  // d_in[2] = use_naive (int) — ignored, both reference paths are identical.
  float* out = (float*)d_out;

  const int total_tiles = NB * HH * (WW / 16);   // 32768 waves
  const int blocks = total_tiles / 8;            // 8 waves (256 thr) per block
  cost_volume_wmma_bf16x2<<<blocks, 256, 0, stream>>>(l_fmap, r_fmap, out);
}
